// HierarchicalEncoder_53506702574126
// MI455X (gfx1250) — matrix operand order repaired and checked by reading, not compiled, not run on hardware
//
// MI455X / gfx1250 — HierarchicalEncoder (ProteinMPNN-style) with bf16 WMMA.
// 77 GFLOP of GEMM work vs ~11us of HBM traffic at 23.3 TB/s: compute-tilted,
// so all matmuls go through v_wmma_f32_16x16x32_bf16 (f32 accumulate).
// A-operands live in LDS in WMMA-fragment-permuted order (2x ds_load_b128 per
// fragment, no repack), B-operands (weights) are pre-swizzled once into
// fragment order (2x global_load_b128 per fragment, coalesced, L2-resident).
// LayerNorm stats / residual / K-sum stay fp32.
#include <hip/hip_runtime.h>

#define BB 2
#define NN 2048
#define KK 32
#define NODE_IN 64
#define EDGE_IN 48
#define HH 128
#define SCALE_F 30.0f
#define EPS_F 1e-6f

typedef __bf16 bf16;
typedef __attribute__((ext_vector_type(16))) __bf16 v16bf;
typedef __attribute__((ext_vector_type(8)))  float  v8f;

// --- index permutations ----------------------------------------------------
// WMMA A-matrix 16x32 layout wants lane<16 to hold K={0..7,16..23} and
// lane>=16 to hold K={8..15,24..31}. Storing K with bits 3<->4 swapped makes
// each lane's 16 elements contiguous (32 bytes).
static __device__ __forceinline__ int permk(int k) {      // swap bits 3,4
  return (k & ~24) | ((k & 8) << 1) | ((k & 16) >> 1);
}
static __device__ __forceinline__ int permj(int j) {      // dword pairs: swap bits 2,3
  return (j & ~12) | ((j & 4) << 1) | ((j & 8) >> 1);
}
// Swizzled B (weight) layout: [kstep][coltile][lane][16 elems] contiguous.
static __device__ __forceinline__ size_t bswz(int k, int c) {
  int kk = k >> 5, w = c >> 4;
  int lane = (c & 15) + (((k >> 3) & 1) << 4);
  return ((size_t)(kk * 8 + w) * 32 + lane) * 16 + ((k & 7) | (((k >> 4) & 1) << 3));
}

// A fragment: 32 contiguous bytes per lane from permuted LDS.
static __device__ __forceinline__ v16bf load_a_frag_sw(const bf16* A, int ldA,
                                                       int mtile, int kbase, int lane) {
  const int row = mtile * 16 + (lane & 15);
  return *(const v16bf*)(A + row * ldA + kbase + ((lane >= 16) ? 16 : 0));
}
// B fragment: 32 contiguous bytes per lane from swizzled weights.
static __device__ __forceinline__ v16bf load_b_frag_sw(const bf16* W, int kk,
                                                       int wv, int lane) {
  return ((const v16bf*)W)[(size_t)(kk * 8 + wv) * 32 + lane];
}

static __device__ __forceinline__ v8f wmma_bf16(v16bf a, v16bf b, v8f c) {
  return __builtin_amdgcn_wmma_f32_16x16x32_bf16(false, a, false, b, (short)0, c,
                                                 false, false);
}

// Permuted column for storing C (bf16) as next GEMM's A operand.
static __device__ __forceinline__ int permcol(int col) {
  return (col & ~31) | (col & 7) | ((col & 16) >> 1) | ((col & 8) << 1);
}

// ---------------------------------------------------------------------------
// One-time conversion: weights fp32 -> bf16 in swizzled fragment order
// (We padded 48->64 K rows with zeros), hS fp32 -> bf16 row-major.
// ---------------------------------------------------------------------------
__global__ void convert_weights(const float* __restrict__ Lw1, const float* __restrict__ Lw2,
                                const float* __restrict__ Lw3, const float* __restrict__ Wv,
                                const float* __restrict__ We,  const float* __restrict__ hS,
                                bf16* __restrict__ w1, bf16* __restrict__ w2,
                                bf16* __restrict__ w3, bf16* __restrict__ wv,
                                bf16* __restrict__ we, bf16* __restrict__ hSb) {
  int i = blockIdx.x * blockDim.x + threadIdx.x;
  if (i < 3 * 512 * HH) {
    int l = i / (512 * HH), rem = i - l * 512 * HH, k = rem / HH, c = rem % HH;
    w1[(size_t)l * 512 * HH + bswz(k, c)] = (bf16)Lw1[i];
  }
  if (i < 3 * HH * HH) {
    int l = i / (HH * HH), rem = i - l * HH * HH, k = rem / HH, c = rem % HH;
    size_t o = (size_t)l * HH * HH + bswz(k, c);
    w2[o] = (bf16)Lw2[i];
    w3[o] = (bf16)Lw3[i];
  }
  if (i < NODE_IN * HH) {
    int k = i / HH, c = i % HH;
    wv[bswz(k, c)] = (bf16)Wv[i];
  }
  if (i < 64 * HH) {
    int k = i / HH, c = i % HH;
    we[bswz(k, c)] = (k < EDGE_IN) ? (bf16)We[k * HH + c] : (bf16)0.0f;
  }
  if (i < BB * NN * HH) hSb[i] = (bf16)hS[i];
}

// ---------------------------------------------------------------------------
// Prep: Y = norm(X @ W + b) for rows of width in_w (<=64, zero-padded to 64).
// One block = 32 rows, 8 waves; wave w owns output cols [16w, 16w+16).
// ---------------------------------------------------------------------------
__global__ void prep_gemm_norm(const float* __restrict__ X, int in_w, int nrows,
                               const bf16* __restrict__ Wb, const float* __restrict__ bias,
                               const float* __restrict__ gain, const float* __restrict__ beta,
                               float* __restrict__ out_f, bf16* __restrict__ out_bf) {
  __shared__ bf16  As[32][64];     // K-permuted
  __shared__ float Cs[32][HH];
  const int tid = threadIdx.x, lane = tid & 31, wv = tid >> 5;
  const int rbase = blockIdx.x * 32;

  for (int i = tid; i < 32 * 64; i += 256) {
    int r = i >> 6, c = i & 63, gr = rbase + r;
    float v = 0.f;
    if (c < in_w && gr < nrows) v = X[(size_t)gr * in_w + c];
    As[r][permk(c)] = (bf16)v;
  }
  __syncthreads();

  const int col = wv * 16 + (lane & 15);
  v8f acc0 = {}, acc1 = {};
  for (int kk = 0; kk < 2; ++kk) {
    v16bf bf_ = load_b_frag_sw(Wb, kk, wv, lane);
    v16bf a0  = load_a_frag_sw(&As[0][0], 64, 0, kk * 32, lane);
    v16bf a1  = load_a_frag_sw(&As[0][0], 64, 1, kk * 32, lane);
    acc0 = wmma_bf16(a0, bf_, acc0);
    acc1 = wmma_bf16(a1, bf_, acc1);
  }
  const float bc = bias[col];
#pragma unroll
  for (int r = 0; r < 8; ++r) {
    int row = r + ((lane >= 16) ? 8 : 0);
    Cs[row][col]      = acc0[r] + bc;
    Cs[row + 16][col] = acc1[r] + bc;
  }
  __syncthreads();

  if (tid < 32) {
    int gr = rbase + tid;
    if (gr < nrows) {
      float s = 0.f;
      for (int j = 0; j < HH; ++j) s += Cs[tid][j];
      float mu = s * (1.0f / HH);
      float var = 0.f;
      for (int j = 0; j < HH; ++j) { float d = Cs[tid][j] - mu; var += d * d; }
      float inv = 1.0f / (sqrtf(var * (1.0f / (HH - 1))) + EPS_F);
      for (int j = 0; j < HH; ++j) {
        float y = gain[j] * (Cs[tid][j] - mu) * inv + beta[j];
        if (out_f)  out_f[(size_t)gr * HH + j]  = y;
        if (out_bf) out_bf[(size_t)gr * HH + j] = (bf16)y;
      }
    }
  }
}

// ---------------------------------------------------------------------------
// One message-passing layer. One block per node (b,n): K=32 rows x 512 cols.
// ---------------------------------------------------------------------------
__global__ void layer_kernel(const bf16* __restrict__ h_bf_in, const float* __restrict__ h_f_in,
                             const bf16* __restrict__ hS_bf,   const bf16* __restrict__ he_bf,
                             const int* __restrict__ E_idx,    const float* __restrict__ mask,
                             const bf16* __restrict__ W1, const float* __restrict__ B1,
                             const bf16* __restrict__ W2, const float* __restrict__ B2,
                             const bf16* __restrict__ W3, const float* __restrict__ B3,
                             const float* __restrict__ Ng, const float* __restrict__ Nb,
                             float* __restrict__ h_f_out, bf16* __restrict__ h_bf_out) {
  __shared__ bf16  As[32][512];   // 32 KB, K-permuted: [h | h_nei | hS_nei | h_e]
  __shared__ bf16  M1[32][HH];    // 8 KB, K-permuted
  __shared__ bf16  M2[32][HH];    // 8 KB, K-permuted
  __shared__ float dh[HH];
  __shared__ float xbuf[HH];
  __shared__ float vmask_s[32];
  __shared__ int   nei_i[32];
  __shared__ float stats[2];

  const int tid = threadIdx.x, lane = tid & 31, wv = tid >> 5;
  const int node = blockIdx.x;          // b*N + n
  const int b = node / NN;

  if (tid < 32) {
    int idx = E_idx[(size_t)node * KK + tid];
    nei_i[tid] = idx;
    vmask_s[tid] = mask[b * NN + idx];
  }
  __syncthreads();

  // Build A rows with dword loads; permute dword position within each
  // 16-dword (32-element) K block so fragments are lane-contiguous.
  {
    const unsigned* hu = (const unsigned*)(h_bf_in + (size_t)node * HH);
    for (int r = wv; r < 32; r += 8) {
      int nidx = nei_i[r];
      const unsigned* hn = (const unsigned*)(h_bf_in + ((size_t)b * NN + nidx) * HH);
      const unsigned* hs = (const unsigned*)(hS_bf   + ((size_t)b * NN + nidx) * HH);
      const unsigned* he = (const unsigned*)(he_bf   + ((size_t)node * KK + r) * HH);
      unsigned* dst = (unsigned*)&As[r][0];
      for (int j = lane; j < 64; j += 32) {
        int pj = permj(j);
        dst[pj]        = hu[j];
        dst[64 + pj]   = hn[j];
        dst[128 + pj]  = hs[j];
        dst[192 + pj]  = he[j];
      }
    }
  }
  __syncthreads();

  const int col  = wv * 16 + (lane & 15);
  const int pcol = permcol(col);

  // GEMM1: [32x512]@[512x128] + b1, relu -> M1 (bf16, permuted)
  {
    v8f acc0 = {}, acc1 = {};
    for (int kk = 0; kk < 16; ++kk) {
      v16bf bf_ = load_b_frag_sw(W1, kk, wv, lane);
      v16bf a0  = load_a_frag_sw(&As[0][0], 512, 0, kk * 32, lane);
      v16bf a1  = load_a_frag_sw(&As[0][0], 512, 1, kk * 32, lane);
      acc0 = wmma_bf16(a0, bf_, acc0);
      acc1 = wmma_bf16(a1, bf_, acc1);
    }
    const float bc = B1[col];
#pragma unroll
    for (int r = 0; r < 8; ++r) {
      int row = r + ((lane >= 16) ? 8 : 0);
      M1[row][pcol]      = (bf16)fmaxf(acc0[r] + bc, 0.f);
      M1[row + 16][pcol] = (bf16)fmaxf(acc1[r] + bc, 0.f);
    }
  }
  __syncthreads();

  // GEMM2: [32x128]@[128x128] + b2, relu -> M2 (bf16, permuted)
  {
    v8f acc0 = {}, acc1 = {};
    for (int kk = 0; kk < 4; ++kk) {
      v16bf bf_ = load_b_frag_sw(W2, kk, wv, lane);
      v16bf a0  = load_a_frag_sw(&M1[0][0], HH, 0, kk * 32, lane);
      v16bf a1  = load_a_frag_sw(&M1[0][0], HH, 1, kk * 32, lane);
      acc0 = wmma_bf16(a0, bf_, acc0);
      acc1 = wmma_bf16(a1, bf_, acc1);
    }
    const float bc = B2[col];
#pragma unroll
    for (int r = 0; r < 8; ++r) {
      int row = r + ((lane >= 16) ? 8 : 0);
      M2[row][pcol]      = (bf16)fmaxf(acc0[r] + bc, 0.f);
      M2[row + 16][pcol] = (bf16)fmaxf(acc1[r] + bc, 0.f);
    }
  }
  __syncthreads();

  // GEMM3: [32x128]@[128x128] + b3, mask per-row, sum over K -> dh[128]
  {
    v8f acc0 = {}, acc1 = {};
    for (int kk = 0; kk < 4; ++kk) {
      v16bf bf_ = load_b_frag_sw(W3, kk, wv, lane);
      v16bf a0  = load_a_frag_sw(&M2[0][0], HH, 0, kk * 32, lane);
      v16bf a1  = load_a_frag_sw(&M2[0][0], HH, 1, kk * 32, lane);
      acc0 = wmma_bf16(a0, bf_, acc0);
      acc1 = wmma_bf16(a1, bf_, acc1);
    }
    const float bc = B3[col];
    float csum = 0.f;
#pragma unroll
    for (int r = 0; r < 8; ++r) {
      int row0 = r + ((lane >= 16) ? 8 : 0);
      csum += (acc0[r] + bc) * vmask_s[row0];
      csum += (acc1[r] + bc) * vmask_s[row0 + 16];
    }
    csum += __shfl_xor(csum, 16);          // combine lane l with lane l+16
    if (lane < 16) dh[col] = csum;
  }
  __syncthreads();

  // Residual + LayerNorm (fp32) + node mask, write next-layer h (f32 + bf16).
  if (tid < HH) xbuf[tid] = h_f_in[(size_t)node * HH + tid] + dh[tid] * (1.0f / SCALE_F);
  __syncthreads();
  if (tid == 0) {
    float s = 0.f;
    for (int j = 0; j < HH; ++j) s += xbuf[j];
    float mu = s * (1.0f / HH);
    float var = 0.f;
    for (int j = 0; j < HH; ++j) { float d = xbuf[j] - mu; var += d * d; }
    stats[0] = mu;
    stats[1] = sqrtf(var * (1.0f / (HH - 1)));
  }
  __syncthreads();
  if (tid < HH) {
    float y = Ng[tid] * (xbuf[tid] - stats[0]) / (stats[1] + EPS_F) + Nb[tid];
    y *= mask[node];
    h_f_out[(size_t)node * HH + tid] = y;
    h_bf_out[(size_t)node * HH + tid] = (bf16)y;
  }
}

// ---------------------------------------------------------------------------
extern "C" void kernel_launch(void* const* d_in, const int* in_sizes, int n_in,
                              void* d_out, int out_size, void* d_ws, size_t ws_size,
                              hipStream_t stream) {
  (void)in_sizes; (void)n_in; (void)out_size; (void)ws_size;
  const float* V     = (const float*)d_in[0];
  const float* E     = (const float*)d_in[1];
  const float* hS    = (const float*)d_in[2];
  const int*   E_idx = (const int*)d_in[3];
  const float* mask  = (const float*)d_in[4];
  const float* Wv_w  = (const float*)d_in[5];
  const float* Wv_b  = (const float*)d_in[6];
  const float* Wv_g  = (const float*)d_in[7];
  const float* Wv_b2 = (const float*)d_in[8];
  const float* We_w  = (const float*)d_in[9];
  const float* We_b  = (const float*)d_in[10];
  const float* We_g  = (const float*)d_in[11];
  const float* We_b2 = (const float*)d_in[12];
  const float* Lw1   = (const float*)d_in[13];
  const float* Lb1   = (const float*)d_in[14];
  const float* Lw2   = (const float*)d_in[15];
  const float* Lb2   = (const float*)d_in[16];
  const float* Lw3   = (const float*)d_in[17];
  const float* Lb3   = (const float*)d_in[18];
  const float* Ln_g  = (const float*)d_in[19];
  const float* Ln_b  = (const float*)d_in[20];

  char* ws = (char*)d_ws;
  size_t off = 0;
  auto carve = [&](size_t bytes) -> void* {
    void* p = ws + off;
    off = (off + bytes + 255) & ~(size_t)255;
    return p;
  };
  bf16* he_bf = (bf16*)carve((size_t)BB * NN * KK * HH * 2);  // 32 MB
  bf16* hS_bf = (bf16*)carve((size_t)BB * NN * HH * 2);
  bf16* h_bfA = (bf16*)carve((size_t)BB * NN * HH * 2);
  bf16* h_bfB = (bf16*)carve((size_t)BB * NN * HH * 2);
  float* h_fA = (float*)carve((size_t)BB * NN * HH * 4);
  float* h_fB = (float*)carve((size_t)BB * NN * HH * 4);
  bf16* w1b = (bf16*)carve((size_t)3 * 512 * HH * 2);
  bf16* w2b = (bf16*)carve((size_t)3 * HH * HH * 2);
  bf16* w3b = (bf16*)carve((size_t)3 * HH * HH * 2);
  bf16* wvb = (bf16*)carve((size_t)NODE_IN * HH * 2);
  bf16* web = (bf16*)carve((size_t)64 * HH * 2);

  convert_weights<<<(BB * NN * HH + 255) / 256, 256, 0, stream>>>(
      Lw1, Lw2, Lw3, Wv_w, We_w, hS, w1b, w2b, w3b, wvb, web, hS_bf);

  prep_gemm_norm<<<BB * NN / 32, 256, 0, stream>>>(
      V, NODE_IN, BB * NN, wvb, Wv_b, Wv_g, Wv_b2, h_fA, h_bfA);

  prep_gemm_norm<<<BB * NN * KK / 32, 256, 0, stream>>>(
      E, EDGE_IN, BB * NN * KK, web, We_b, We_g, We_b2, nullptr, he_bf);

  // Layer 0: A -> B
  layer_kernel<<<BB * NN, 256, 0, stream>>>(
      h_bfA, h_fA, hS_bf, he_bf, E_idx, mask,
      w1b + (size_t)0 * 512 * HH, Lb1 + 0 * HH,
      w2b + (size_t)0 * HH * HH,  Lb2 + 0 * HH,
      w3b + (size_t)0 * HH * HH,  Lb3 + 0 * HH,
      Ln_g + 0 * HH, Ln_b + 0 * HH, h_fB, h_bfB);
  // Layer 1: B -> A
  layer_kernel<<<BB * NN, 256, 0, stream>>>(
      h_bfB, h_fB, hS_bf, he_bf, E_idx, mask,
      w1b + (size_t)1 * 512 * HH, Lb1 + 1 * HH,
      w2b + (size_t)1 * HH * HH,  Lb2 + 1 * HH,
      w3b + (size_t)1 * HH * HH,  Lb3 + 1 * HH,
      Ln_g + 1 * HH, Ln_b + 1 * HH, h_fA, h_bfA);
  // Layer 2: A -> d_out (fp32)
  layer_kernel<<<BB * NN, 256, 0, stream>>>(
      h_bfA, h_fA, hS_bf, he_bf, E_idx, mask,
      w1b + (size_t)2 * 512 * HH, Lb1 + 2 * HH,
      w2b + (size_t)2 * HH * HH,  Lb2 + 2 * HH,
      w3b + (size_t)2 * HH * HH,  Lb3 + 2 * HH,
      Ln_g + 2 * HH, Ln_b + 2 * HH, (float*)d_out, h_bfB);
}
